// HierarchicalAffinityLoss_28690381537966
// MI455X (gfx1250) — matrix operand order repaired and checked by reading, not compiled, not run on hardware
//
#include <hip/hip_runtime.h>
#include <math.h>

// ---------------------------------------------------------------------------
// HierarchicalAffinityLoss on gfx1250 (MI455X, wave32, WMMA)
//
// dist = 1 - Nn @ Nn^T computed with V_WMMA_F32_16X16X32_F16 using an
// fp32 -> (f16 hi + f16 lo) split: a.b ~= ah*bh + ah*bl + al*bh  (fp32 acc).
// Each wave owns a 32x32 output register tile (2x2 WMMA subtiles, 4 f32
// accumulators) so every A/B fragment load feeds two WMMAs. Masked
// hardest-positive max / hardest-negative min fused into the GEMM epilogue
// via order-preserving-uint atomics; the 64M-entry distance matrix is never
// materialized.
// ---------------------------------------------------------------------------

typedef __attribute__((ext_vector_type(16))) _Float16 v16h;
typedef __attribute__((ext_vector_type(8)))  _Float16 v8h;
typedef __attribute__((ext_vector_type(8)))  float    v8f;

__device__ __constant__ int kFamTable[6] = {0, 1, 2, 1, 3, 3};
#define MARGIN 0.3f

// order-preserving map: float total order <-> unsigned total order
static __device__ __forceinline__ unsigned f2ord(float f) {
  unsigned u = __float_as_uint(f);
  return (u & 0x80000000u) ? ~u : (u | 0x80000000u);
}
static __device__ __forceinline__ float ord2f(unsigned k) {
  unsigned u = (k & 0x80000000u) ? (k & 0x7fffffffu) : ~k;
  return __uint_as_float(u);
}

static __device__ __forceinline__ v16h combine8(v8h a, v8h b) {
  return __builtin_shufflevector(a, b, 0, 1, 2, 3, 4, 5, 6, 7,
                                 8, 9, 10, 11, 12, 13, 14, 15);
}

// ---------------------------------------------------------------------------
// Kernel 1: family lookup, family histogram, sentinel init (single block)
// ---------------------------------------------------------------------------
__global__ void prep_kernel(const int* __restrict__ labels,
                            int* __restrict__ fam,
                            unsigned* __restrict__ posKey,
                            unsigned* __restrict__ negKey,
                            int* __restrict__ famCount, int B) {
  __shared__ int cnt[8];
  if (threadIdx.x < 8) cnt[threadIdx.x] = 0;
  __syncthreads();
  for (int i = threadIdx.x; i < B; i += blockDim.x) {
    int lab = labels[i];
    int f = (lab >= 0 && lab < 6) ? kFamTable[lab] : -1;
    fam[i] = f;
    if (f >= 0) atomicAdd(&cnt[f], 1);
    posKey[i] = f2ord(-2.0f);   // sentinel for hardest positive (max)
    negKey[i] = f2ord(4.0f);    // sentinel for hardest negative (min)
  }
  __syncthreads();
  if (threadIdx.x < 8) famCount[threadIdx.x] = cnt[threadIdx.x];
}

// ---------------------------------------------------------------------------
// Kernel 2: L2-normalize rows, split into f16 hi/lo. One wave per row.
// ---------------------------------------------------------------------------
__global__ void normalize_split_kernel(const float* __restrict__ emb,
                                       _Float16* __restrict__ nhi,
                                       _Float16* __restrict__ nlo,
                                       int B, int D) {
  int row = blockIdx.x * (blockDim.x >> 5) + (threadIdx.x >> 5);
  if (row >= B) return;
  int lane = threadIdx.x & 31;
  const float* e = emb + (size_t)row * D;
  float ss = 0.0f;
  for (int k = lane; k < D; k += 32) { float v = e[k]; ss += v * v; }
  for (int m = 16; m >= 1; m >>= 1) ss += __shfl_xor(ss, m);
  float rn = 1.0f / sqrtf(ss);
  _Float16* ho = nhi + (size_t)row * D;
  _Float16* lo = nlo + (size_t)row * D;
  for (int k = lane; k < D; k += 32) {
    float v = e[k] * rn;
    _Float16 h = (_Float16)v;
    ho[k] = h;
    lo[k] = (_Float16)(v - (float)h);
  }
}

// ---------------------------------------------------------------------------
// Kernel 3: fused WMMA GEMM (N @ N^T) + masked row max/min reduction.
// 256 threads = 8 waves; block tile 64(i) x 128(j); each wave a 32x32 tile
// made of 2x2 16x16 WMMA subtiles (4 accumulators, full A/B fragment reuse).
// ---------------------------------------------------------------------------
__global__ void __launch_bounds__(256)
wmma_triplet_kernel(const _Float16* __restrict__ nhi,
                    const _Float16* __restrict__ nlo,
                    const int* __restrict__ fam,
                    unsigned* __restrict__ posKey,
                    unsigned* __restrict__ negKey,
                    int B, int D) {
  const int tid  = threadIdx.x;
  const int wid  = tid >> 5;
  const int lane = tid & 31;
  const int wi = wid & 1;          // 0..1  -> 32-row strip
  const int wj = wid >> 1;         // 0..3  -> 32-col strip
  const int i0 = blockIdx.y * 64 + wi * 32;
  const int j0 = blockIdx.x * 128 + wj * 32;
  const int g   = lane >> 4;       // half-wave group
  const int l16 = lane & 15;

  // A fragments (rows i0+16s): lane holds row l16, K-halves {8g..8g+7, 16+8g..}
  const _Float16* aHi0 = nhi + (size_t)(i0 + l16) * D;
  const _Float16* aLo0 = nlo + (size_t)(i0 + l16) * D;
  const _Float16* aHi1 = nhi + (size_t)(i0 + 16 + l16) * D;
  const _Float16* aLo1 = nlo + (size_t)(i0 + 16 + l16) * D;
  // B fragments (rows j0+16t of N = columns of N^T): lane holds row l16, K-halves {8g..8g+7, 16+8g..}
  const _Float16* bHi0 = nhi + (size_t)(j0 + l16) * D;
  const _Float16* bLo0 = nlo + (size_t)(j0 + l16) * D;
  const _Float16* bHi1 = nhi + (size_t)(j0 + 16 + l16) * D;
  const _Float16* bLo1 = nlo + (size_t)(j0 + 16 + l16) * D;

  v8f acc00 = {}, acc01 = {}, acc10 = {}, acc11 = {};

  for (int k = 0; k < D; k += 32) {
    const int ka = k + 8 * g;        // A chunk offsets
    const int kb = k + 8 * g;        // B chunk offset
    v16h ah0 = combine8(*(const v8h*)(aHi0 + ka), *(const v8h*)(aHi0 + ka + 16));
    v16h al0 = combine8(*(const v8h*)(aLo0 + ka), *(const v8h*)(aLo0 + ka + 16));
    v16h ah1 = combine8(*(const v8h*)(aHi1 + ka), *(const v8h*)(aHi1 + ka + 16));
    v16h al1 = combine8(*(const v8h*)(aLo1 + ka), *(const v8h*)(aLo1 + ka + 16));
    v16h bh0 = combine8(*(const v8h*)(bHi0 + kb), *(const v8h*)(bHi0 + kb + 16));
    v16h bl0 = combine8(*(const v8h*)(bLo0 + kb), *(const v8h*)(bLo0 + kb + 16));
    v16h bh1 = combine8(*(const v8h*)(bHi1 + kb), *(const v8h*)(bHi1 + kb + 16));
    v16h bl1 = combine8(*(const v8h*)(bLo1 + kb), *(const v8h*)(bLo1 + kb + 16));
    // fp32-split product per subtile: ah*bh + ah*bl + al*bh, fp32 accumulate
    acc00 = __builtin_amdgcn_wmma_f32_16x16x32_f16(false, ah0, false, bh0, (short)0, acc00, false, false);
    acc00 = __builtin_amdgcn_wmma_f32_16x16x32_f16(false, ah0, false, bl0, (short)0, acc00, false, false);
    acc00 = __builtin_amdgcn_wmma_f32_16x16x32_f16(false, al0, false, bh0, (short)0, acc00, false, false);
    acc01 = __builtin_amdgcn_wmma_f32_16x16x32_f16(false, ah0, false, bh1, (short)0, acc01, false, false);
    acc01 = __builtin_amdgcn_wmma_f32_16x16x32_f16(false, ah0, false, bl1, (short)0, acc01, false, false);
    acc01 = __builtin_amdgcn_wmma_f32_16x16x32_f16(false, al0, false, bh1, (short)0, acc01, false, false);
    acc10 = __builtin_amdgcn_wmma_f32_16x16x32_f16(false, ah1, false, bh0, (short)0, acc10, false, false);
    acc10 = __builtin_amdgcn_wmma_f32_16x16x32_f16(false, ah1, false, bl0, (short)0, acc10, false, false);
    acc10 = __builtin_amdgcn_wmma_f32_16x16x32_f16(false, al1, false, bh0, (short)0, acc10, false, false);
    acc11 = __builtin_amdgcn_wmma_f32_16x16x32_f16(false, ah1, false, bh1, (short)0, acc11, false, false);
    acc11 = __builtin_amdgcn_wmma_f32_16x16x32_f16(false, ah1, false, bl1, (short)0, acc11, false, false);
    acc11 = __builtin_amdgcn_wmma_f32_16x16x32_f16(false, al1, false, bh1, (short)0, acc11, false, false);
  }

  // Epilogue: C layout -> lane L, reg r holds (M = r + 8*g, N = l16).
  // For fixed (s, r), the 16 lanes of one half-wave share row i and span 16 j.
  const int fj0 = fam[j0 + l16];
  const int fj1 = fam[j0 + 16 + l16];
#pragma unroll
  for (int s = 0; s < 2; ++s) {
#pragma unroll
    for (int r = 0; r < 8; ++r) {
      int i  = i0 + 16 * s + 8 * g + r;
      int fi = fam[i];
      float p = -2.0f, n = 4.0f;
      {
        int j = j0 + l16;
        float d = 1.0f - (s == 0 ? acc00[r] : acc10[r]);
        if (fi >= 0 && fi == fj0 && i != j) p = fmaxf(p, d);
        if (fi >= 0 && fj0 >= 0 && fi != fj0) n = fminf(n, d);
      }
      {
        int j = j0 + 16 + l16;
        float d = 1.0f - (s == 0 ? acc01[r] : acc11[r]);
        if (fi >= 0 && fi == fj1 && i != j) p = fmaxf(p, d);
        if (fi >= 0 && fj1 >= 0 && fi != fj1) n = fminf(n, d);
      }
      for (int m = 1; m < 16; m <<= 1) {
        p = fmaxf(p, __shfl_xor(p, m));
        n = fminf(n, __shfl_xor(n, m));
      }
      if (l16 == 0) {
        if (p > -2.0f) atomicMax(&posKey[i], f2ord(p));
        if (n < 4.0f)  atomicMin(&negKey[i], f2ord(n));
      }
    }
  }
}

// ---------------------------------------------------------------------------
// Kernel 4: finalize -> scalar loss (single block)
// ---------------------------------------------------------------------------
__global__ void finalize_kernel(const unsigned* __restrict__ posKey,
                                const unsigned* __restrict__ negKey,
                                const int* __restrict__ fam,
                                const int* __restrict__ famCount,
                                float* __restrict__ out, int B) {
  __shared__ float ssum[256];
  __shared__ int   scnt[256];
  int totalValid = 0;
  for (int f = 0; f < 8; ++f) totalValid += famCount[f];
  float sum = 0.0f;
  int cnt = 0;
  for (int i = threadIdx.x; i < B; i += blockDim.x) {
    int f = fam[i];
    bool hasSame = (f >= 0) && (famCount[f] >= 2);
    bool hasDiff = (f >= 0) && (totalValid - famCount[f] >= 1);
    if (hasSame && hasDiff) {
      float dp = ord2f(posKey[i]);
      float dn = ord2f(negKey[i]);
      float t = dp - dn + MARGIN;
      sum += (t > 0.0f) ? t : 0.0f;
      cnt += 1;
    }
  }
  ssum[threadIdx.x] = sum;
  scnt[threadIdx.x] = cnt;
  __syncthreads();
  for (int s = 128; s > 0; s >>= 1) {
    if ((int)threadIdx.x < s) {
      ssum[threadIdx.x] += ssum[threadIdx.x + s];
      scnt[threadIdx.x] += scnt[threadIdx.x + s];
    }
    __syncthreads();
  }
  if (threadIdx.x == 0) {
    int c = scnt[0] > 0 ? scnt[0] : 1;
    out[0] = ssum[0] / (float)c;
  }
}

// ---------------------------------------------------------------------------
// Launch
// ---------------------------------------------------------------------------
extern "C" void kernel_launch(void* const* d_in, const int* in_sizes, int n_in,
                              void* d_out, int out_size, void* d_ws, size_t ws_size,
                              hipStream_t stream) {
  const float* emb    = (const float*)d_in[0];
  const int*   labels = (const int*)d_in[1];
  float* out = (float*)d_out;

  const int B = in_sizes[1];             // 8192
  const int D = in_sizes[0] / B;         // 256

  // workspace carve-up (~8.1 MB)
  _Float16* nhi = (_Float16*)d_ws;
  _Float16* nlo = nhi + (size_t)B * D;
  unsigned* posKey = (unsigned*)(nlo + (size_t)B * D);
  unsigned* negKey = posKey + B;
  int* fam = (int*)(negKey + B);
  int* famCount = fam + B;

  prep_kernel<<<1, 1024, 0, stream>>>(labels, fam, posKey, negKey, famCount, B);

  // one wave per row, 8 waves per block
  normalize_split_kernel<<<(B + 7) / 8, 256, 0, stream>>>(emb, nhi, nlo, B, D);

  dim3 grid(B / 128, B / 64);            // (64, 128) blocks of 8 waves
  wmma_triplet_kernel<<<grid, 256, 0, stream>>>(nhi, nlo, fam, posKey, negKey, B, D);

  finalize_kernel<<<1, 256, 0, stream>>>(posKey, negKey, fam, famCount, out, B);
}
